// MDT_29772713296063
// MI455X (gfx1250) — matrix operand order repaired and checked by reading, not compiled, not run on hardware
//
#include <hip/hip_runtime.h>

// ---------------------------------------------------------------------------
// Fused windowed-attention for MI455X (gfx1250, wave32, WMMA bf16).
// One 128-thread block (4 waves) per window of 64 tokens x 96 channels.
// All intermediates (qkv, attn logits, probs, O) live in LDS; HBM traffic is
// only x (201 MB) in + out (201 MB) -> memory-bound, ~17us roofline.
// ---------------------------------------------------------------------------

typedef __bf16 bf16_t;
typedef __attribute__((ext_vector_type(16))) __bf16 v16bf;
typedef __attribute__((ext_vector_type(8)))  __bf16 v8bf;
typedef __attribute__((ext_vector_type(8)))  float  v8f;

#define N_WIN     8192
#define N_TOK     64
#define C_DIM     96
#define TBL       15
#define N_IMG     32
#define XS        104   // padded LDS stride (bf16 elems) for 96-col matrices
#define PS        72    // padded LDS stride (bf16 elems) for 64-col matrices

// ---- workspace layout (bytes) ----
// [0      , 55296) : qkv_w as bf16 (288*96)
// [55296  , 73728) : proj_w as bf16 (96*96)
// [73728  , 73792) : phi[15]  f32
// [73792  , 75712) : theta[32*15] f32
#define WS_PROJW_B  55296
#define WS_PHI_B    73728
#define WS_THETA_B  73792

// ===========================================================================
// Prep: bf16 weight copies + 15-entry bias tables (phi constant, theta/image)
// ===========================================================================
__global__ void prep_kernel(const float* __restrict__ qkv_w,
                            const float* __restrict__ proj_w,
                            const float* __restrict__ a_p,
                            const float* __restrict__ b_p,
                            const float* __restrict__ a_r,
                            const float* __restrict__ b_r,
                            const float* __restrict__ theta_max,
                            bf16_t* __restrict__ qkvw_b,
                            bf16_t* __restrict__ projw_b,
                            float* __restrict__ phi,
                            float* __restrict__ theta) {
  int t = blockIdx.x * blockDim.x + threadIdx.x;
  int stride = gridDim.x * blockDim.x;
  for (int i = t; i < 288 * 96; i += stride) qkvw_b[i] = (bf16_t)qkv_w[i];
  for (int i = t; i < 96 * 96;  i += stride) projw_b[i] = (bf16_t)proj_w[i];

  const float TWO_PI_OVER_128 = 6.28318530717958647692f / 128.0f;
  if (t < TBL) {                       // azimuth bias: a_p[d%15]*cos + b_p*sin
    int d = t - 7;
    int idx = d < 0 ? d + TBL : d;     // python-style mod
    float az = (float)d * TWO_PI_OVER_128;
    phi[t] = a_p[idx] * __cosf(az) + b_p[idx] * __sinf(az);
  }
  if (t < N_IMG * TBL) {               // radius bias: per-image theta_max
    int img = t / TBL;
    int d = (t % TBL) - 7;
    int idx = d < 0 ? d + TBL : d;
    float rn = (float)d * theta_max[img] * (1.0f / 128.0f);
    theta[t] = a_r[idx] * __cosf(rn) + b_r[idx] * __sinf(rn);
  }
}

// ===========================================================================
// WMMA operand loaders (ISA 7.12.2 layouts).
// A (16x32 bf16): lane = M row; VGPR0-3 hold K = k0+8*half .. +7,
//                 VGPR4-7 hold K = k0+16+8*half .. +7   -> two 16B chunks.
// B (32x16 bf16): lane = N column (a row of the row-major source matrix);
//                 same K order as A -> two 16B chunks.
// ===========================================================================
static __device__ __forceinline__
v16bf load_tile_A(const bf16_t* base, int stride, int row0, int k0,
                  int ln, int half) {
  const bf16_t* p = base + (row0 + ln) * stride;
  v8bf lo = *(const v8bf*)(p + k0 + 8 * half);
  v8bf hi = *(const v8bf*)(p + k0 + 16 + 8 * half);
  return __builtin_shufflevector(lo, hi, 0,1,2,3,4,5,6,7,8,9,10,11,12,13,14,15);
}

static __device__ __forceinline__
v16bf load_tile_B(const bf16_t* base, int stride, int n0, int k0,
                  int ln, int half) {
  const bf16_t* p = base + (n0 + ln) * stride + k0 + 8 * half;
  v8bf lo = *(const v8bf*)(p);
  v8bf hi = *(const v8bf*)(p + 16);
  return __builtin_shufflevector(lo, hi, 0,1,2,3,4,5,6,7,8,9,10,11,12,13,14,15);
}

static __device__ __forceinline__ v8f wmma_bf16(v16bf a, v16bf b, v8f c) {
  // 8 args: (neg_a, A, neg_b, B, c_mod, C, reuse_a, reuse_b)
  return __builtin_amdgcn_wmma_f32_16x16x32_bf16(false, a, false, b,
                                                 (short)0, c, false, false);
}

// ===========================================================================
// Main fused kernel: one block per window.
// ===========================================================================
__global__ __launch_bounds__(128)
void attn_window_kernel(const float* __restrict__ x,
                        const bf16_t* __restrict__ qkvw,
                        const float* __restrict__ qkv_b,
                        const bf16_t* __restrict__ projw,
                        const float* __restrict__ proj_b,
                        const float* __restrict__ phi,
                        const float* __restrict__ theta,
                        float* __restrict__ out) {
  __shared__ bf16_t sX[N_TOK * XS];      // x (bf16); reused as O after phase 3
  __shared__ bf16_t sQ[N_TOK * XS];      // Q, pre-scaled by 96^-0.5
  __shared__ bf16_t sK[N_TOK * XS];      // K row-major
  __shared__ bf16_t sVt[C_DIM * PS];     // V transposed: [channel][token]
  __shared__ bf16_t sP[N_TOK * PS];      // softmax probs
  __shared__ float  sBias[TBL * TBL];    // combined bias: theta[di]+phi[dj]

  const int w    = blockIdx.x;
  const int img  = w >> 8;               // 256 windows per image
  const int tid  = threadIdx.x;
  const int lane = tid & 31;
  const int wave = tid >> 5;
  const int half = lane >> 4;
  const int ln   = lane & 15;
  const int m0   = wave * 16;            // this wave's token-row tile

  // ---- combined bias table + load x (f32 -> bf16) -------------------------
  if (tid < TBL * TBL) {
    int di = tid / TBL, dj = tid % TBL;
    sBias[tid] = theta[img * TBL + di] + phi[dj];
  }
  const float* xg = x + (size_t)w * (N_TOK * C_DIM);
  for (int i = tid; i < N_TOK * C_DIM; i += 128) {
    int r = i / C_DIM, c = i - r * C_DIM;
    sX[r * XS + c] = (bf16_t)xg[i];
  }
  __syncthreads();

  // ---- phase 1: qkv = x @ qkv_w^T + qkv_b  (M=64, N=288, K=96) ------------
  {
    const float scale = 0.1020620726159658f;  // 96^-0.5 applied to Q
    for (int nt = 0; nt < 18; ++nt) {
      v8f acc = {};
      for (int kt = 0; kt < 3; ++kt) {
        v16bf a = load_tile_A(sX, XS, m0, kt * 32, ln, half);
        v16bf b = load_tile_B(qkvw, C_DIM, nt * 16, kt * 32, ln, half);
        acc = wmma_bf16(a, b, acc);
      }
      int cj = nt * 16 + ln;               // output column 0..287
      float bias = qkv_b[cj];
      if (cj < 96) {                        // Q
        for (int r = 0; r < 8; ++r)
          sQ[(m0 + r + 8 * half) * XS + cj] = (bf16_t)((acc[r] + bias) * scale);
      } else if (cj < 192) {                // K
        int c = cj - 96;
        for (int r = 0; r < 8; ++r)
          sK[(m0 + r + 8 * half) * XS + c] = (bf16_t)(acc[r] + bias);
      } else {                              // V -> transposed store
        int c = cj - 192;
        for (int r = 0; r < 8; ++r)
          sVt[c * PS + (m0 + r + 8 * half)] = (bf16_t)(acc[r] + bias);
      }
    }
  }
  __syncthreads();

  // ---- phase 2: attn = Q@K^T + bias, softmax -> sP ------------------------
  {
    v8f acc[4];
    for (int nt = 0; nt < 4; ++nt) {
      v8f a_ = {};
      for (int kt = 0; kt < 3; ++kt) {
        v16bf av = load_tile_A(sQ, XS, m0, kt * 32, ln, half);
        v16bf bv = load_tile_B(sK, XS, nt * 16, kt * 32, ln, half);
        a_ = wmma_bf16(av, bv, a_);
      }
      acc[nt] = a_;
    }
    // relative-position bias: token n=(i,j) 8x8 grid, lookup by (di,dj)
    for (int nt = 0; nt < 4; ++nt) {
      int mcol = nt * 16 + ln;
      int mi = mcol >> 3, mj = mcol & 7;
      for (int r = 0; r < 8; ++r) {
        int row = m0 + r + 8 * half;
        int di = (row >> 3) - mi + 7;
        int dj = (row & 7) - mj + 7;
        acc[nt][r] += sBias[di * TBL + dj];
      }
    }
    // softmax per token row: row r+8*half spread over one 16-lane group x 4 regs
    for (int r = 0; r < 8; ++r) {
      float mx = fmaxf(fmaxf(acc[0][r], acc[1][r]), fmaxf(acc[2][r], acc[3][r]));
      for (int o = 1; o < 16; o <<= 1) mx = fmaxf(mx, __shfl_xor(mx, o, 32));
      float e0 = __expf(acc[0][r] - mx), e1 = __expf(acc[1][r] - mx);
      float e2 = __expf(acc[2][r] - mx), e3 = __expf(acc[3][r] - mx);
      float s = e0 + e1 + e2 + e3;
      for (int o = 1; o < 16; o <<= 1) s += __shfl_xor(s, o, 32);
      float inv = 1.0f / s;
      int row = m0 + r + 8 * half;
      sP[row * PS +  0 + ln] = (bf16_t)(e0 * inv);
      sP[row * PS + 16 + ln] = (bf16_t)(e1 * inv);
      sP[row * PS + 32 + ln] = (bf16_t)(e2 * inv);
      sP[row * PS + 48 + ln] = (bf16_t)(e3 * inv);
    }
  }
  __syncthreads();

  // ---- phase 3: O = P @ V  (M=64, N=96, K=64); store into sX (reuse) ------
  {
    for (int nt = 0; nt < 6; ++nt) {
      v8f acc = {};
      for (int kt = 0; kt < 2; ++kt) {
        v16bf av = load_tile_A(sP, PS, m0, kt * 32, ln, half);
        v16bf bv = load_tile_B(sVt, PS, nt * 16, kt * 32, ln, half);
        acc = wmma_bf16(av, bv, acc);
      }
      int c = nt * 16 + ln;
      for (int r = 0; r < 8; ++r)
        sX[(m0 + r + 8 * half) * XS + c] = (bf16_t)acc[r];
    }
  }
  __syncthreads();

  // ---- phase 4: out = O @ proj_w^T + proj_b  (M=64, N=96, K=96) -----------
  {
    float* og = out + (size_t)w * (N_TOK * C_DIM);
    for (int nt = 0; nt < 6; ++nt) {
      v8f acc = {};
      for (int kt = 0; kt < 3; ++kt) {
        v16bf av = load_tile_A(sX, XS, m0, kt * 32, ln, half);
        v16bf bv = load_tile_B(projw, C_DIM, nt * 16, kt * 32, ln, half);
        acc = wmma_bf16(av, bv, acc);
      }
      int c = nt * 16 + ln;
      float pb = proj_b[c];
      for (int r = 0; r < 8; ++r)
        og[(m0 + r + 8 * half) * C_DIM + c] = acc[r] + pb;
    }
  }
}

// ===========================================================================
extern "C" void kernel_launch(void* const* d_in, const int* in_sizes, int n_in,
                              void* d_out, int out_size, void* d_ws, size_t ws_size,
                              hipStream_t stream) {
  (void)in_sizes; (void)n_in; (void)out_size; (void)ws_size;

  const float* x         = (const float*)d_in[0];
  const float* theta_max = (const float*)d_in[1];
  const float* a_p       = (const float*)d_in[2];
  const float* b_p       = (const float*)d_in[3];
  const float* a_r       = (const float*)d_in[4];
  const float* b_r       = (const float*)d_in[5];
  const float* qkv_w     = (const float*)d_in[6];
  const float* qkv_b     = (const float*)d_in[7];
  const float* proj_w    = (const float*)d_in[8];
  const float* proj_b    = (const float*)d_in[9];
  float* out = (float*)d_out;

  char* ws = (char*)d_ws;
  bf16_t* qkvw_b = (bf16_t*)ws;
  bf16_t* projw_b = (bf16_t*)(ws + WS_PROJW_B);
  float*  phi     = (float*)(ws + WS_PHI_B);
  float*  theta   = (float*)(ws + WS_THETA_B);

  prep_kernel<<<16, 256, 0, stream>>>(qkv_w, proj_w, a_p, b_p, a_r, b_r,
                                      theta_max, qkvw_b, projw_b, phi, theta);
  attn_window_kernel<<<N_WIN, 128, 0, stream>>>(x, qkvw_b, qkv_b, projw_b,
                                                proj_b, phi, theta, out);
}
